// Moca_net_20392504722038
// MI455X (gfx1250) — matrix operand order repaired and checked by reading, not compiled, not run on hardware
//
#include <hip/hip_runtime.h>
#include <math.h>

typedef __attribute__((ext_vector_type(16))) _Float16 v16h;
typedef __attribute__((ext_vector_type(8)))  float    v8f;

#define LSEQ 32768
#define NBATCH 2
#define NCH 48
#define DM_ 24
#define DIN_ 48
#define DS_ 8
#define NCHUNK 128
#define CHUNKL 256

__device__ __forceinline__ int voxel_of(int iter, int l) {
    int a = l >> 10, b = (l >> 5) & 31, c = l & 31;
    if (iter == 0) return l;                        // (d,h,w) = (a,b,c)
    if (iter == 1) return (c << 10) + (a << 5) + b; // seq order h,w,d
    return (b << 10) + (c << 5) + a;                // seq order w,d,h
}

__device__ __forceinline__ float silu_f(float x) { return x / (1.0f + __expf(-x)); }
__device__ __forceinline__ float softplus_f(float x) {
    return (x > 20.0f) ? x : log1pf(__expf(x));
}
// ISA 16-bit A-matrix 16x32 layout: half-slot p (0..15) -> K
__device__ __forceinline__ int kmapA(int p, int lane) {
    return ((p < 8) ? p : p + 8) + ((lane >> 4) << 3);
}

// ---------------- gather + layernorm (+ reverse for backward stream) ----------
__global__ void gather_ln_kernel(const float* __restrict__ src,
                                 const float* __restrict__ g,
                                 const float* __restrict__ bb,
                                 float* __restrict__ uf, float* __restrict__ ub,
                                 int iter) {
    __shared__ float sv[64 * 48];
    __shared__ float smu[64], srs[64];
    const int b  = blockIdx.y;
    const int l0 = blockIdx.x * 64;
    const int tid = threadIdx.x;
    for (int idx = tid; idx < 64 * 48; idx += 256) {
        int c = idx >> 6, p = idx & 63;
        int v = voxel_of(iter, l0 + p);
        sv[p * 48 + c] = src[(b * NCH + c) * LSEQ + v];
    }
    __syncthreads();
    if (tid < 64) {
        float mu = 0.f;
        #pragma unroll
        for (int c = 0; c < 48; ++c) mu += sv[tid * 48 + c];
        mu *= (1.0f / 48.0f);
        float var = 0.f;
        #pragma unroll
        for (int c = 0; c < 48; ++c) { float d = sv[tid * 48 + c] - mu; var += d * d; }
        var *= (1.0f / 48.0f);
        smu[tid] = mu;
        srs[tid] = rsqrtf(var + 1e-5f);
    }
    __syncthreads();
    for (int idx = tid; idx < 64 * 24; idx += 256) {
        int p = idx / 24, cc = idx % 24;
        int l = l0 + p;
        float mu = smu[p], rs = srs[p];
        float x0 = (sv[p * 48 + cc] - mu) * rs * g[cc] + bb[cc];
        uf[((size_t)b * LSEQ + l) * DM_ + cc] = x0;
        float x1 = (sv[p * 48 + 24 + cc] - mu) * rs * g[24 + cc] + bb[24 + cc];
        ub[((size_t)b * LSEQ + (LSEQ - 1 - l)) * DM_ + cc] = x1;
    }
}

// -------- in-projection: u[B,L,24] @ in_w^T[24->96] -> xz[B,L,96]  (WMMA) -----
// 192 threads = 6 waves; block covers 64 seq positions; wave w owns cols 16w..16w+15.
// D tiles staged through LDS, then coalesced float4 stores.
__global__ void inproj_kernel(const float* __restrict__ u,
                              const float* __restrict__ w_in, // [96,24]
                              float* __restrict__ xz) {       // [B,L,96]
    __shared__ _Float16 su[64 * 32];
    __shared__ __align__(16) float sxz[64 * 96];
    const int b  = blockIdx.y;
    const int l0 = blockIdx.x * 64;
    const int tid = threadIdx.x;
    const float* ublk = u + ((size_t)b * LSEQ + l0) * DM_;
    for (int idx = tid; idx < 64 * 32; idx += 192) {
        int p = idx >> 5, k = idx & 31;
        su[idx] = (k < DM_) ? (_Float16)ublk[p * DM_ + k] : (_Float16)0.0f;
    }
    __syncthreads();
    const int wave = tid >> 5, lane = tid & 31;
    const int nn = wave * 16 + (lane & 15);
    // B fragment (K x N = 32 x 16), zero-padded K>=24
    v16h bf;
    const int kbaseB = (lane >> 4) * 8;
    #pragma unroll
    for (int p = 0; p < 16; ++p) {
        int k = kbaseB + ((p < 8) ? p : p + 8);
        bf[p] = (k < DM_) ? (_Float16)w_in[nn * DM_ + k] : (_Float16)0.0f;
    }
    const int m = lane & 15;
    const int mhi = (lane >> 4) << 3;
    // mt = 0 (no reuse hint on the first of the identical-instruction run)
    {
        v16h af;
        #pragma unroll
        for (int p = 0; p < 16; ++p)
            af[p] = su[m * 32 + kmapA(p, lane)];
        v8f cf = {};
        cf = __builtin_amdgcn_wmma_f32_16x16x32_f16(false, af, false, bf,
                                                    (short)0, cf, false, false);
        #pragma unroll
        for (int v = 0; v < 8; ++v)
            sxz[(v + mhi) * 96 + nn] = cf[v];
    }
    // mt = 1..3: identical instruction, same B operand -> reuse_b hint
    #pragma unroll
    for (int mt = 1; mt < 4; ++mt) {
        v16h af;
        #pragma unroll
        for (int p = 0; p < 16; ++p)
            af[p] = su[(mt * 16 + m) * 32 + kmapA(p, lane)];
        v8f cf = {};
        cf = __builtin_amdgcn_wmma_f32_16x16x32_f16(false, af, false, bf,
                                                    (short)0, cf, false, true);
        #pragma unroll
        for (int v = 0; v < 8; ++v)
            sxz[(mt * 16 + v + mhi) * 96 + nn] = cf[v];
    }
    __syncthreads();
    float4* __restrict__ d4 = (float4*)(xz + ((size_t)b * LSEQ + l0) * 96);
    const float4* __restrict__ s4 = (const float4*)sxz;
    for (int idx = tid; idx < (64 * 96) / 4; idx += 192) d4[idx] = s4[idx];
}

// ------------- conv(4-tap causal)+silu, x-proj(48->18), dt(2->48, softplus) ---
__global__ void conv_xproj_dt_kernel(const float* __restrict__ xz,  // [B,L,96]
                                     const float* __restrict__ cw,  // [48,4]
                                     const float* __restrict__ cb,  // [48]
                                     const float* __restrict__ xw,  // [18,48]
                                     const float* __restrict__ dtw, // [48,2]
                                     const float* __restrict__ dtb, // [48]
                                     float* __restrict__ xc_out,
                                     float* __restrict__ dt_out,
                                     float* __restrict__ Bm,
                                     float* __restrict__ Cm) {
    int gi = blockIdx.x * 256 + threadIdx.x;
    int b = gi >> 15, l = gi & (LSEQ - 1);
    const float* xp = xz + (size_t)b * LSEQ * 96;
    float acc[48];
    #pragma unroll
    for (int d = 0; d < 48; ++d) acc[d] = cb[d];
    #pragma unroll
    for (int k = 0; k < 4; ++k) {
        int ls = l + k - 3;
        if (ls >= 0) {
            const float4* row4 = (const float4*)(xp + (size_t)ls * 96);
            #pragma unroll
            for (int q = 0; q < 12; ++q) {
                float4 r = row4[q];
                acc[4 * q + 0] += r.x * cw[(4 * q + 0) * 4 + k];
                acc[4 * q + 1] += r.y * cw[(4 * q + 1) * 4 + k];
                acc[4 * q + 2] += r.z * cw[(4 * q + 2) * 4 + k];
                acc[4 * q + 3] += r.w * cw[(4 * q + 3) * 4 + k];
            }
        }
    }
    #pragma unroll
    for (int d = 0; d < 48; ++d) acc[d] = silu_f(acc[d]);
    size_t lidx = (size_t)b * LSEQ + l;
    #pragma unroll
    for (int d = 0; d < 48; ++d) xc_out[lidx * 48 + d] = acc[d];
    float dtin0 = 0.f, dtin1 = 0.f;
    #pragma unroll
    for (int d = 0; d < 48; ++d) { dtin0 += acc[d] * xw[0 * 48 + d];
                                   dtin1 += acc[d] * xw[1 * 48 + d]; }
    #pragma unroll
    for (int j = 0; j < 8; ++j) {
        float sB = 0.f, sC = 0.f;
        #pragma unroll
        for (int d = 0; d < 48; ++d) { sB += acc[d] * xw[(2 + j) * 48 + d];
                                       sC += acc[d] * xw[(10 + j) * 48 + d]; }
        Bm[lidx * 8 + j] = sB;
        Cm[lidx * 8 + j] = sC;
    }
    #pragma unroll
    for (int d = 0; d < 48; ++d) {
        float t = dtin0 * dtw[d * 2 + 0] + dtin1 * dtw[d * 2 + 1] + dtb[d];
        dt_out[lidx * 48 + d] = softplus_f(t);
    }
}

// ---------------- scan phase A: per-chunk (prod dA, partial h) ----------------
__global__ void scan_partial_kernel(const float* __restrict__ dt,
                                    const float* __restrict__ xc,
                                    const float* __restrict__ Bm,
                                    const float* __restrict__ Alog, // [384]
                                    float* __restrict__ Aprod,
                                    float* __restrict__ Hpart) {
    __shared__ float sdt[8 * 48], sxc[8 * 48], sbm[8 * 8];
    const int chunk = blockIdx.x, b = blockIdx.y, t = threadIdx.x;
    const int d = t >> 3, s = t & 7;
    const float A = -__expf(Alog[t]);
    float Ap = 1.0f, H = 0.0f;
    const size_t base = (size_t)b * LSEQ + chunk * CHUNKL;
    for (int gblk = 0; gblk < CHUNKL / 8; ++gblk) {
        for (int idx = t; idx < 8 * 104; idx += 384) {
            int r = idx / 104, j = idx % 104;
            size_t row = base + gblk * 8 + r;
            if (j < 48)      sdt[r * 48 + j]      = dt[row * 48 + j];
            else if (j < 96) sxc[r * 48 + j - 48] = xc[row * 48 + j - 48];
            else             sbm[r * 8 + j - 96]  = Bm[row * 8 + j - 96];
        }
        __syncthreads();
        #pragma unroll
        for (int r = 0; r < 8; ++r) {
            float dtv = sdt[r * 48 + d];
            float dA  = __expf(dtv * A);
            float dBx = dtv * sbm[r * 8 + s] * sxc[r * 48 + d];
            Ap *= dA;
            H = dA * H + dBx;
        }
        __syncthreads();
    }
    size_t ci = ((size_t)b * NCHUNK + chunk) * 384 + t;
    Aprod[ci] = Ap;
    Hpart[ci] = H;
}

// ---------------- scan phase B: combine chunk partials -> h0 per chunk --------
__global__ void scan_combine_kernel(const float* __restrict__ Aprod,
                                    const float* __restrict__ Hpart,
                                    float* __restrict__ H0) {
    const int b = blockIdx.x, t = threadIdx.x;
    float h = 0.0f;
    for (int k = 0; k < NCHUNK; ++k) {
        size_t ci = ((size_t)b * NCHUNK + k) * 384 + t;
        H0[ci] = h;
        h = Aprod[ci] * h + Hpart[ci];
    }
}

// ------- scan phase C: replay with h0, y = h.C, +xc*Dp, gate with silu(z) -----
__global__ void scan_final_kernel(const float* __restrict__ dt,
                                  const float* __restrict__ xc,
                                  const float* __restrict__ Bm,
                                  const float* __restrict__ Cm,
                                  const float* __restrict__ xz,   // z = xz[...,48:96]
                                  const float* __restrict__ Alog,
                                  const float* __restrict__ Dpar, // [48]
                                  const float* __restrict__ H0,
                                  float* __restrict__ yg) {
    __shared__ float sdt[8 * 48], sxc[8 * 48], sz[8 * 48], sbm[8 * 8], scm[8 * 8];
    const int chunk = blockIdx.x, b = blockIdx.y, t = threadIdx.x;
    const int d = t >> 3, s = t & 7;
    const float A = -__expf(Alog[t]);
    const float Dpv = Dpar[d];
    const size_t base = (size_t)b * LSEQ + chunk * CHUNKL;
    float H = H0[((size_t)b * NCHUNK + chunk) * 384 + t];
    for (int gblk = 0; gblk < CHUNKL / 8; ++gblk) {
        for (int idx = t; idx < 8 * 160; idx += 384) {
            int r = idx / 160, j = idx % 160;
            size_t row = base + gblk * 8 + r;
            if (j < 48)       sdt[r * 48 + j]       = dt[row * 48 + j];
            else if (j < 96)  sxc[r * 48 + j - 48]  = xc[row * 48 + j - 48];
            else if (j < 144) sz [r * 48 + j - 96]  = xz[row * 96 + 48 + (j - 96)];
            else if (j < 152) sbm[r * 8 + j - 144]  = Bm[row * 8 + j - 144];
            else              scm[r * 8 + j - 152]  = Cm[row * 8 + j - 152];
        }
        __syncthreads();
        #pragma unroll
        for (int r = 0; r < 8; ++r) {
            float dtv = sdt[r * 48 + d];
            float dA  = __expf(dtv * A);
            float dBx = dtv * sbm[r * 8 + s] * sxc[r * 48 + d];
            H = dA * H + dBx;
            float prod = H * scm[r * 8 + s];
            #pragma unroll
            for (int off = 1; off < 8; off <<= 1) prod += __shfl_xor(prod, off, 32);
            if (s == 0) {
                float yv = prod + sxc[r * 48 + d] * Dpv;
                float zv = sz[r * 48 + d];
                yg[(base + gblk * 8 + r) * 48 + d] = yv * silu_f(zv);
            }
        }
        __syncthreads();
    }
}

// ---------------- out-projection: yg[B,L,48] @ ow^T[48->24] (WMMA, 2 k-steps) -
// 128 threads = 4 waves; block covers 32 positions; wave=(mtile,ntile).
// D tiles staged through LDS, then coalesced float4 stores.
__global__ void outproj_kernel(const float* __restrict__ yg,
                               const float* __restrict__ ow, // [24,48]
                               float* __restrict__ op) {     // [B,L,24]
    __shared__ _Float16 sy[32 * 64];
    __shared__ __align__(16) float so[32 * 24];
    const int b  = blockIdx.y;
    const int l0 = blockIdx.x * 32;
    const int tid = threadIdx.x;
    const float* yblk = yg + ((size_t)b * LSEQ + l0) * 48;
    for (int idx = tid; idx < 32 * 64; idx += 128) {
        int p = idx >> 6, k = idx & 63;
        sy[idx] = (k < 48) ? (_Float16)yblk[p * 48 + k] : (_Float16)0.0f;
    }
    __syncthreads();
    const int wave = tid >> 5, lane = tid & 31;
    const int mt = wave >> 1, nt = wave & 1;
    const int nn = nt * 16 + (lane & 15);
    v16h bf0, bf1;
    const int kbaseB = (lane >> 4) * 8;
    #pragma unroll
    for (int p = 0; p < 16; ++p) {
        int k0 = kbaseB + ((p < 8) ? p : p + 8), k1 = 32 + k0;
        bf0[p] = (nn < 24 && k0 < 48) ? (_Float16)ow[nn * 48 + k0] : (_Float16)0.0f;
        bf1[p] = (nn < 24 && k1 < 48) ? (_Float16)ow[nn * 48 + k1] : (_Float16)0.0f;
    }
    const int m = lane & 15;
    const int mhi = (lane >> 4) << 3;
    v16h af0, af1;
    #pragma unroll
    for (int p = 0; p < 16; ++p) {
        int km = kmapA(p, lane);
        af0[p] = sy[(mt * 16 + m) * 64 + km];
        af1[p] = sy[(mt * 16 + m) * 64 + 32 + km];
    }
    v8f cf = {};
    cf = __builtin_amdgcn_wmma_f32_16x16x32_f16(false, af0, false, bf0, (short)0, cf, false, false);
    cf = __builtin_amdgcn_wmma_f32_16x16x32_f16(false, af1, false, bf1, (short)0, cf, false, false);
    if (nn < 24) {
        #pragma unroll
        for (int v = 0; v < 8; ++v)
            so[(mt * 16 + v + mhi) * 24 + nn] = cf[v];
    }
    __syncthreads();
    float4* __restrict__ d4 = (float4*)(op + ((size_t)b * LSEQ + l0) * DM_);
    const float4* __restrict__ s4 = (const float4*)so;
    for (int idx = tid; idx < (32 * 24) / 4; idx += 128) d4[idx] = s4[idx];
}

// ---------------- scatter + residual (+ final +x) -----------------------------
__global__ void scatter_kernel(const float* __restrict__ src,
                               const float* __restrict__ opF,
                               const float* __restrict__ opB,
                               const float* __restrict__ xin,
                               float* __restrict__ dst,
                               int iter, int add_x) {
    const int b  = blockIdx.y;
    const int l0 = blockIdx.x * 64;
    const int tid = threadIdx.x;
    for (int idx = tid; idx < 64 * 48; idx += 256) {
        int p = idx / 48, c = idx % 48;
        int l = l0 + p;
        int v = voxel_of(iter, l);
        size_t gidx = ((size_t)b * NCH + c) * LSEQ + v;
        float contrib = (c < 24)
            ? opF[((size_t)b * LSEQ + l) * DM_ + c]
            : opB[((size_t)b * LSEQ + (LSEQ - 1 - l)) * DM_ + (c - 24)];
        float o = src[gidx] + contrib;
        if (add_x) o += xin[gidx];
        dst[gidx] = o;
    }
}

extern "C" void kernel_launch(void* const* d_in, const int* in_sizes, int n_in,
                              void* d_out, int out_size, void* d_ws, size_t ws_size,
                              hipStream_t stream) {
    const float* x     = (const float*)d_in[0];
    const float* ln_g  = (const float*)d_in[1];
    const float* ln_b  = (const float*)d_in[2];
    const float* in_w  = (const float*)d_in[3];
    const float* cw    = (const float*)d_in[4];
    const float* cb    = (const float*)d_in[5];
    const float* xw    = (const float*)d_in[6];
    const float* dtw   = (const float*)d_in[7];
    const float* dtb   = (const float*)d_in[8];
    const float* Alog  = (const float*)d_in[9];
    const float* Dp    = (const float*)d_in[10];
    const float* ow    = (const float*)d_in[11];

    const size_t NF  = (size_t)NBATCH * LSEQ;      // 65536 positions
    float* w = (float*)d_ws;
    size_t o = 0;
    float* curA  = w + o; o += NF * 48;
    float* curB  = w + o; o += NF * 48;
    float* uf    = w + o; o += NF * DM_;
    float* ub    = w + o; o += NF * DM_;
    float* xzbuf = w + o; o += NF * 96;
    float* xcb   = w + o; o += NF * 48;
    float* dtbuf = w + o; o += NF * 48;
    float* ygb   = w + o; o += NF * 48;
    float* Bmb   = w + o; o += NF * 8;
    float* Cmb   = w + o; o += NF * 8;
    float* opF   = w + o; o += NF * DM_;
    float* opB   = w + o; o += NF * DM_;
    float* Aprod = w + o; o += (size_t)NBATCH * NCHUNK * 384;
    float* Hpart = w + o; o += (size_t)NBATCH * NCHUNK * 384;
    float* H0b   = w + o; o += (size_t)NBATCH * NCHUNK * 384;

    for (int i = 0; i < 3; ++i) {
        const float* src = (i == 0) ? x : ((i == 1) ? curA : curB);
        float* dst = (i == 0) ? curA : ((i == 1) ? curB : (float*)d_out);

        gather_ln_kernel<<<dim3(LSEQ / 64, NBATCH), 256, 0, stream>>>(
            src, ln_g + i * 48, ln_b + i * 48, uf, ub, i);

        for (int dir = 0; dir < 2; ++dir) {
            const int e = 2 * i + dir;
            const float* u  = dir ? ub : uf;
            float* op       = dir ? opB : opF;

            inproj_kernel<<<dim3(LSEQ / 64, NBATCH), 192, 0, stream>>>(
                u, in_w + (size_t)e * 96 * 24, xzbuf);

            conv_xproj_dt_kernel<<<dim3((NBATCH * LSEQ) / 256), 256, 0, stream>>>(
                xzbuf, cw + (size_t)e * 48 * 4, cb + (size_t)e * 48,
                xw + (size_t)e * 18 * 48, dtw + (size_t)e * 48 * 2,
                dtb + (size_t)e * 48, xcb, dtbuf, Bmb, Cmb);

            scan_partial_kernel<<<dim3(NCHUNK, NBATCH), 384, 0, stream>>>(
                dtbuf, xcb, Bmb, Alog + (size_t)e * 384, Aprod, Hpart);

            scan_combine_kernel<<<dim3(NBATCH), 384, 0, stream>>>(Aprod, Hpart, H0b);

            scan_final_kernel<<<dim3(NCHUNK, NBATCH), 384, 0, stream>>>(
                dtbuf, xcb, Bmb, Cmb, xzbuf, Alog + (size_t)e * 384,
                Dp + (size_t)e * 48, H0b, ygb);

            outproj_kernel<<<dim3(LSEQ / 32, NBATCH), 128, 0, stream>>>(
                ygb, ow + (size_t)e * 24 * 48, op);
        }

        scatter_kernel<<<dim3(LSEQ / 64, NBATCH), 256, 0, stream>>>(
            src, opF, opB, x, dst, i, (i == 2) ? 1 : 0);
    }
}
